// Block_2422361555046
// MI455X (gfx1250) — matrix operand order repaired and checked by reading, not compiled, not run on hardware
//
#include <hip/hip_runtime.h>
#include <math.h>

// ---------------------------------------------------------------------------
// ConvNeXt-ish block on gfx1250 (MI455X), wave32 + WMMA bf16.
//   y = SAT piecewise part (per-channel 14x14 G table)            [VALU]
//     + x_planes @ T^T (shared kpe operator, dense 4096x4096)     [WMMA bf16]
//   then LN -> GEMM1(+b1, exact GELU) -> GRN -> GEMM2(+b2+beta@w2+residual)
// B-side matrices are materialized K-minor so A and B fragments are each
// 2x ds_load_b128.  GEMM tile staging uses the Tensor Data Mover (TDM);
// k_pc stages its SAT plane with async global->LDS loads.
// ---------------------------------------------------------------------------

typedef __attribute__((ext_vector_type(16))) __bf16 bf16x16;
typedef __attribute__((ext_vector_type(8)))  float  floatx8;
typedef __attribute__((ext_vector_type(4)))  unsigned int u32x4;
typedef __attribute__((ext_vector_type(8)))  int i32x8;
typedef __attribute__((ext_vector_type(4)))  int i32x4;

#define AS1 __attribute__((address_space(1)))
#define AS3 __attribute__((address_space(3)))

#if defined(__has_builtin)
#  if __has_builtin(__builtin_amdgcn_global_load_async_to_lds_b128) && \
      __has_builtin(__builtin_amdgcn_s_wait_asynccnt)
#    define HAVE_ASYNC 1
#  endif
#  if __has_builtin(__builtin_amdgcn_tensor_load_to_lds) && \
      __has_builtin(__builtin_amdgcn_s_wait_tensorcnt)
#    define HAVE_TDM 1
#  endif
#endif
#ifndef HAVE_ASYNC
#  define HAVE_ASYNC 0
#endif
#ifndef HAVE_TDM
#  define HAVE_TDM 0
#endif

#define CDIM 384
#define HID  1536
#define BATCH 8
#define HW   4096   // 64*64
#define NPIX 32768  // BATCH*HW

// ------------------------------- small utils -------------------------------

__global__ void k_f32_to_bf16(const float* __restrict__ s, __bf16* __restrict__ d,
                              size_t n) {
  size_t i = (size_t)blockIdx.x * blockDim.x + threadIdx.x;
  size_t stride = (size_t)gridDim.x * blockDim.x;
  for (; i < n; i += stride) d[i] = (__bf16)s[i];
}

// dst[n*R + k] = (bf16) src[k*C + n]   (transpose-convert, src is RxC f32)
__global__ void k_transpose_bf16(const float* __restrict__ src, __bf16* __restrict__ d,
                                 int R, int C) {
  size_t n_elems = (size_t)R * C;
  size_t i = (size_t)blockIdx.x * blockDim.x + threadIdx.x;
  size_t stride = (size_t)gridDim.x * blockDim.x;
  for (; i < n_elems; i += stride) {
    int n = (int)(i / R), k = (int)(i % R);
    d[i] = (__bf16)src[(size_t)k * C + n];
  }
}

// bv[c] = sum_f grn_beta[f] * w2[f, c]
__global__ void k_bv(const float* __restrict__ beta, const float* __restrict__ w2,
                     float* __restrict__ bv) {
  __shared__ float red[256];
  int c = blockIdx.x, t = threadIdx.x;
  float s = 0.f;
  for (int f = t; f < HID; f += 256) s += beta[f] * w2[(size_t)f * CDIM + c];
  red[t] = s; __syncthreads();
  for (int o = 128; o > 0; o >>= 1) { if (t < o) red[t] += red[t + o]; __syncthreads(); }
  if (t == 0) bv[c] = red[0];
}

// -------------------- peripheral-kernel structure helpers ------------------

__global__ void k_buildG(const float* __restrict__ wcomp, float* __restrict__ G) {
  int idx = blockIdx.x * blockDim.x + threadIdx.x;
  if (idx >= CDIM * 196) return;
  int c = idx / 196, st = idx % 196, s = st / 14, t = st % 14;
  auto W = [&](int a, int b) -> float {
    return (a >= 0 && a < 13 && b >= 0 && b < 13) ? wcomp[(size_t)c * 169 + a * 13 + b] : 0.f;
  };
  G[idx] = W(s - 1, t - 1) - W(s, t - 1) - W(s - 1, t) + W(s, t);
}

// K-minor kpe operator: Tt[q][p] = kpe[(i2-i)+25][(j2-j)+25]
__global__ void k_build_tt(const float* __restrict__ kpe, __bf16* __restrict__ Tt) {
  size_t idx = (size_t)blockIdx.x * 256 + threadIdx.x;   // 16.7M
  int q = (int)(idx >> 12), p = (int)(idx & 4095);
  int i = q >> 6, j = q & 63, i2 = p >> 6, j2 = p & 63;
  int di = i2 - i + 25, dj = j2 - j + 25;
  float v = (di >= 0 && di < 51 && dj >= 0 && dj < 51) ? kpe[di * 51 + dj] : 0.f;
  Tt[idx] = (__bf16)v;
}

// per-plane summed-area table (64x64)
__global__ void k_sat(const float* __restrict__ x, float* __restrict__ ss) {
  __shared__ float t[64][65];
  int plane = blockIdx.x, tid = threadIdx.x;   // 64 threads
  const float* xp = x + (size_t)plane * HW;
  for (int r = 0; r < 64; ++r) t[r][tid] = xp[r * 64 + tid];
  __syncthreads();
  float run = 0.f;
  for (int j = 0; j < 64; ++j) { run += t[tid][j]; t[tid][j] = run; }
  __syncthreads();
  run = 0.f;
  for (int i = 0; i < 64; ++i) { run += t[i][tid]; t[i][tid] = run; }
  __syncthreads();
  float* sp = ss + (size_t)plane * HW;
  for (int r = 0; r < 64; ++r) sp[r * 64 + tid] = t[r][tid];
}

// y += sum_{s,t} G[c][s][t] * SAT(i+m[s], j+m[t])   (196 taps, clamped)
__global__ __launch_bounds__(256)
void k_pc(const float* __restrict__ SS, const float* __restrict__ G,
          float* __restrict__ y) {
  __shared__ float S[64][64];
  __shared__ float g[196];
  int plane = blockIdx.x, c = plane % CDIM, tid = threadIdx.x;
  const float* sp = SS + (size_t)plane * HW;
#if HAVE_ASYNC
  for (int r = 0; r < 4; ++r) {
    int e = (tid + 256 * r) * 4;        // 16B per async op, 4096 floats total
    __builtin_amdgcn_global_load_async_to_lds_b128(
        (AS1 i32x4*)(sp + e), (AS3 i32x4*)(&S[0][0] + e), 0, 0);
  }
  if (tid < 196) g[tid] = G[(size_t)c * 196 + tid];
  __builtin_amdgcn_s_wait_asynccnt(0);
#else
  for (int r = 0; r < 16; ++r) {
    int idx = tid + 256 * r;
    S[idx >> 6][idx & 63] = sp[idx];
  }
  if (tid < 196) g[tid] = G[(size_t)c * 196 + tid];
#endif
  __syncthreads();
  const int moff[14] = {-26, -25, -16, -8, -4, -2, -1, 0, 1, 3, 7, 15, 24, 25};
  float* yp = y + (size_t)plane * HW;
  for (int it = 0; it < 16; ++it) {
    int pix = tid + 256 * it;
    int i = pix >> 6, j = pix & 63;
    float acc = yp[pix];
#pragma unroll
    for (int s = 0; s < 14; ++s) {
      int r = i + moff[s];
      if (r < 0) continue;
      const float* row = S[r > 63 ? 63 : r];
      float gacc = 0.f;
#pragma unroll
      for (int t = 0; t < 14; ++t) {
        int cc = j + moff[t];
        if (cc < 0) continue;
        gacc += g[s * 14 + t] * row[cc > 63 ? 63 : cc];
      }
      acc += gacc;
    }
    yp[pix] = acc;
  }
}

// ------------------------------ LayerNorm ----------------------------------
__global__ __launch_bounds__(256)
void k_ln(const float* __restrict__ y, const float* __restrict__ lnw,
          const float* __restrict__ lnb, __bf16* __restrict__ hln) {
  __shared__ float S[CDIM * 33];
  __shared__ float ps[8][32], ps2[8][32];
  __shared__ float muS[32], rsS[32];
  int gb = blockIdx.x;
  int b = gb >> 7;
  int p0 = (gb & 127) * 32;
  int tid = threadIdx.x;
  for (int r = 0; r < 48; ++r) {
    int idx = tid + 256 * r;
    int c = idx >> 5, pp = idx & 31;
    S[c * 33 + pp] = y[((size_t)b * CDIM + c) * HW + p0 + pp];
  }
  __syncthreads();
  int pp = tid & 31, seg = tid >> 5;
  float s = 0.f, s2 = 0.f;
  for (int c = seg * 48; c < seg * 48 + 48; ++c) {
    float v = S[c * 33 + pp]; s += v; s2 += v * v;
  }
  ps[seg][pp] = s; ps2[seg][pp] = s2;
  __syncthreads();
  if (tid < 32) {
    float a = 0.f, a2 = 0.f;
    for (int k = 0; k < 8; ++k) { a += ps[k][tid]; a2 += ps2[k][tid]; }
    float m = a * (1.f / CDIM);
    float var = a2 * (1.f / CDIM) - m * m;
    muS[tid] = m; rsS[tid] = rsqrtf(var + 1e-6f);
  }
  __syncthreads();
  for (int r = 0; r < 48; ++r) {
    int idx = tid + 256 * r;
    int p2 = idx / CDIM, c = idx - p2 * CDIM;
    float v = (S[c * 33 + p2] - muS[p2]) * rsS[p2] * lnw[c] + lnb[c];
    hln[((size_t)(b * HW) + p0 + p2) * CDIM + c] = (__bf16)v;
  }
}

// --------------------------------- GRN -------------------------------------

__global__ void k_grn_reduce(const __bf16* __restrict__ H1, float* __restrict__ gx2) {
  int f = blockIdx.x * 256 + threadIdx.x;
  int b = blockIdx.y;
  int p0 = blockIdx.z * 256;
  float s = 0.f;
  for (int p = p0; p < p0 + 256; ++p) {
    float v = (float)H1[((size_t)b * HW + p) * HID + f];
    s += v * v;
  }
  atomicAdd(&gx2[b * HID + f], s);
}

__global__ void k_grn_scale(const float* __restrict__ gx2, const float* __restrict__ gamma,
                            float* __restrict__ sfac) {
  __shared__ float red[256];
  __shared__ float meanS;
  int b = blockIdx.x, t = threadIdx.x;
  float s = 0.f;
  for (int f = t; f < HID; f += 256) s += sqrtf(gx2[b * HID + f]);
  red[t] = s; __syncthreads();
  for (int o = 128; o > 0; o >>= 1) { if (t < o) red[t] += red[t + o]; __syncthreads(); }
  if (t == 0) meanS = red[0] * (1.f / HID);
  __syncthreads();
  for (int f = t; f < HID; f += 256) {
    float gx = sqrtf(gx2[b * HID + f]);
    sfac[b * HID + f] = 1.f + gamma[f] * (gx / (meanS + 1e-6f));
  }
}

__global__ void k_grn_apply(__bf16* __restrict__ H1, const float* __restrict__ sfac) {
  size_t n = (size_t)NPIX * HID;
  size_t i = (size_t)blockIdx.x * blockDim.x + threadIdx.x;
  size_t stride = (size_t)gridDim.x * blockDim.x;
  for (; i < n; i += stride) {
    int f = (int)(i % HID);
    int b = (int)(i / ((size_t)HW * HID));
    H1[i] = (__bf16)((float)H1[i] * sfac[b * HID + f]);
  }
}

// ------------------------------ WMMA GEMM ----------------------------------
// C(MxN) = A(MxK,row-major) @ B, B given K-minor: Bt[n][k], ld = Kd.
// 128x128x32 tiles, 8 wave32 waves (4x2), per wave 2x4 WMMA 16x16 tiles.
// LDS tiles K-minor, stride 40 elems (64B row + 16B pad): 16B-aligned b128,
// 20-word bank stride.  TDM D# pad: 4 DWORDs after every 16 DWORDs == same.

__device__ __forceinline__ int ldsoff(int rc, int k) { return rc * 40 + k; }

#if HAVE_TDM
// Issue a TDM 2-D tile load: 32(k) x 128(rows) bf16 tile of a rows x Kd
// K-minor matrix -> LDS at lds_off with stride-40 padding.
// D# layout per CDNA5 ISA 8.3/8.4 (group0 128b, group1 256b).
__device__ __forceinline__ void tdm_load_tile(const __bf16* gsrc, unsigned lds_off,
                                              int rows_total, int Kd) {
  unsigned long long ga = (unsigned long long)(size_t)gsrc;
  u32x4 g0;
  g0.x = 1u;                                    // count=1 (valid user D#)
  g0.y = lds_off;                               // lds_addr (bytes)
  g0.z = (unsigned)ga;                          // global_addr[31:0]
  g0.w = (unsigned)(ga >> 32) | (2u << 30);     // global_addr[56:32] | type=2
  i32x8 g1;
  g1[0] = (1 << 16) |                           // data_size = 2 bytes
          (1 << 20) |                           // pad_enable
          (3 << 22) |                           // pad_interval: 16 DWORDs
          (3 << 25);                            // pad_amount: 4 DWORDs
  g1[1] = (int)(((unsigned)Kd & 0xFFFFu) << 16);            // tensor_dim0[15:0]
  g1[2] = (int)(((unsigned)Kd >> 16) |
                (((unsigned)rows_total & 0xFFFFu) << 16));   // d0[31:16]|d1[15:0]
  g1[3] = (int)(((unsigned)rows_total >> 16) | (32u << 16)); // d1[31:16]|tile_dim0
  g1[4] = 128;                                  // tile_dim1 = 128 rows
  g1[5] = Kd;                                   // tensor_dim0_stride[31:0]
  g1[6] = 0;                                    // d0_stride[47:32] | d1_stride lo
  g1[7] = 0;
  i32x4 z4 = {0, 0, 0, 0};
  i32x8 z8 = {0, 0, 0, 0, 0, 0, 0, 0};
  __builtin_amdgcn_tensor_load_to_lds(g0, g1, z4, z4, z8, 0);
}
#endif

template <int EPI>
__global__ __launch_bounds__(256)
void gemm_bf16(const __bf16* __restrict__ A, const __bf16* __restrict__ Bt,
               float* __restrict__ Cf, __bf16* __restrict__ Cb,
               int M, int N, int Kd,
               const float* __restrict__ bias, const float* __restrict__ bias2,
               const float* __restrict__ xres) {
  __shared__ __bf16 As[128 * 40];   // [m][k], stride 40
  __shared__ __bf16 Bs[128 * 40];   // [n][k], stride 40

  const int tid = threadIdx.x;
  const int lane = tid & 31;
  const int wave = tid >> 5;
  const int wm = (wave & 3) * 32;
  const int wn = (wave >> 2) * 64;
  const int bm = blockIdx.y * 128;
  const int bn = blockIdx.x * 128;

  floatx8 acc[2][4];
  const floatx8 zacc = {0.f, 0.f, 0.f, 0.f, 0.f, 0.f, 0.f, 0.f};
#pragma unroll
  for (int i = 0; i < 2; ++i)
#pragma unroll
    for (int j = 0; j < 4; ++j) acc[i][j] = zacc;

  const int sr = tid >> 1;            // staging row (A) / col (B): 0..127
  const int sc = (tid & 1) << 4;      // 0 / 16
  const int kbA = (lane & 16) ? 8 : 0;
  const int kbB = (lane & 16) ? 8 : 0;
#if HAVE_TDM
  // low 32 bits of a generic LDS pointer == LDS byte offset (flat aperture)
  const unsigned asOff = (unsigned)(size_t)(void*)As;
  const unsigned bsOff = (unsigned)(size_t)(void*)Bs;
#endif

  for (int k0 = 0; k0 < Kd; k0 += 32) {
#if HAVE_TDM
    if (tid == 0) {                   // TDM ignores EXEC; one issue per block
      tdm_load_tile(A  + (size_t)bm * Kd + k0, asOff, M, Kd);
      tdm_load_tile(Bt + (size_t)bn * Kd + k0, bsOff, N, Kd);
      __builtin_amdgcn_s_wait_tensorcnt(0);
    }
    __syncthreads();
#else
    const __bf16* pa = A  + (size_t)(bm + sr) * Kd + (k0 + sc);
    const __bf16* pb = Bt + (size_t)(bn + sr) * Kd + (k0 + sc);
#if HAVE_ASYNC
    __builtin_amdgcn_global_load_async_to_lds_b128(
        (AS1 i32x4*)pa, (AS3 i32x4*)&As[ldsoff(sr, sc)], 0, 0);
    __builtin_amdgcn_global_load_async_to_lds_b128(
        (AS1 i32x4*)(pa + 8), (AS3 i32x4*)&As[ldsoff(sr, sc) + 8], 0, 0);
    __builtin_amdgcn_global_load_async_to_lds_b128(
        (AS1 i32x4*)pb, (AS3 i32x4*)&Bs[ldsoff(sr, sc)], 0, 0);
    __builtin_amdgcn_global_load_async_to_lds_b128(
        (AS1 i32x4*)(pb + 8), (AS3 i32x4*)&Bs[ldsoff(sr, sc) + 8], 0, 0);
    __builtin_amdgcn_s_wait_asynccnt(0);
#else
    *(uint4*)&As[ldsoff(sr, sc)]     = *(const uint4*)pa;
    *(uint4*)&As[ldsoff(sr, sc) + 8] = *(const uint4*)(pa + 8);
    *(uint4*)&Bs[ldsoff(sr, sc)]     = *(const uint4*)pb;
    *(uint4*)&Bs[ldsoff(sr, sc) + 8] = *(const uint4*)(pb + 8);
#endif
    if (k0 + 32 < Kd) {                            // -> global_prefetch_b8
      __builtin_prefetch(pa + 32, 0, 0);
      __builtin_prefetch(pb + 32, 0, 0);
    }
    __syncthreads();
#endif

    union Frag { bf16x16 v; uint4 u4[2]; };
    Frag af[2], bf[4];
#pragma unroll
    for (int mi = 0; mi < 2; ++mi) {
      int row = wm + mi * 16 + (lane & 15);
      af[mi].u4[0] = *(const uint4*)&As[ldsoff(row, kbA)];
      af[mi].u4[1] = *(const uint4*)&As[ldsoff(row, 16 + kbA)];
    }
#pragma unroll
    for (int ni = 0; ni < 4; ++ni) {
      int col = wn + ni * 16 + (lane & 15);
      bf[ni].u4[0] = *(const uint4*)&Bs[ldsoff(col, kbB)];
      bf[ni].u4[1] = *(const uint4*)&Bs[ldsoff(col, 16 + kbB)];
    }
#pragma unroll
    for (int mi = 0; mi < 2; ++mi)
#pragma unroll
      for (int ni = 0; ni < 4; ++ni)
        acc[mi][ni] = __builtin_amdgcn_wmma_f32_16x16x32_bf16(
            false, af[mi].v, false, bf[ni].v, (short)0, acc[mi][ni], false, false);
    __syncthreads();
  }

  const int rowhi = (lane & 16) ? 8 : 0;
  const int cl = lane & 15;
#pragma unroll
  for (int mi = 0; mi < 2; ++mi)
#pragma unroll
    for (int ni = 0; ni < 4; ++ni)
#pragma unroll
      for (int r = 0; r < 8; ++r) {
        int row = bm + wm + mi * 16 + rowhi + r;
        int col = bn + wn + ni * 16 + cl;
        float d = acc[mi][ni][r];
        if (EPI == 0) {
          Cf[(size_t)row * N + col] = d;
        } else if (EPI == 1) {
          float v = d + bias[col];
          float gelu = 0.5f * v * (1.0f + erff(v * 0.70710678118654752f));
          Cb[(size_t)row * N + col] = (__bf16)gelu;
        } else {
          int b = row >> 12, pix = row & 4095;
          size_t o = ((size_t)b * CDIM + col) * HW + pix;   // NCHW scatter
          Cf[o] = d + bias[col] + bias2[col] + xres[o];
        }
      }
}

// ------------------------------- launcher ----------------------------------

extern "C" void kernel_launch(void* const* d_in, const int* in_sizes, int n_in,
                              void* d_out, int out_size, void* d_ws, size_t ws_size,
                              hipStream_t stream) {
  (void)in_sizes; (void)n_in; (void)out_size; (void)ws_size;
  const float* x   = (const float*)d_in[0];
  const float* wc  = (const float*)d_in[1];
  const float* kpe = (const float*)d_in[2];
  const float* lnw = (const float*)d_in[3];
  const float* lnb = (const float*)d_in[4];
  const float* w1  = (const float*)d_in[5];
  const float* b1  = (const float*)d_in[6];
  const float* gg  = (const float*)d_in[7];
  const float* gb  = (const float*)d_in[8];
  const float* w2  = (const float*)d_in[9];
  const float* b2  = (const float*)d_in[10];
  float* out = (float*)d_out;

  char* ws = (char*)d_ws;
  size_t off = 0;
  auto alloc = [&](size_t bytes) -> void* {
    void* p = ws + off;
    off += (bytes + 255) & ~(size_t)255;
    return p;
  };
  char*   R1  = (char*)alloc((size_t)NPIX * HID * 2);      // Tt / H1 overlap
  __bf16* Tt  = (__bf16*)R1;
  __bf16* H1  = (__bf16*)R1;
  char*   R2  = (char*)alloc((size_t)NPIX * CDIM * 2);     // xb / hln overlap
  __bf16* xb  = (__bf16*)R2;
  __bf16* hln = (__bf16*)R2;
  float* SS   = (float*)alloc((size_t)BATCH * CDIM * HW * 4);
  float* y    = (float*)alloc((size_t)BATCH * CDIM * HW * 4);
  float* G    = (float*)alloc((size_t)CDIM * 196 * 4);
  __bf16* w1t = (__bf16*)alloc((size_t)CDIM * HID * 2);    // [n=HID][k=CDIM]
  __bf16* w2t = (__bf16*)alloc((size_t)HID * CDIM * 2);    // [n=CDIM][k=HID]
  float* gx2  = (float*)alloc((size_t)BATCH * HID * 4);
  float* sfac = (float*)alloc((size_t)BATCH * HID * 4);
  float* bv   = (float*)alloc((size_t)CDIM * 4);

  (void)hipMemsetAsync(gx2, 0, (size_t)BATCH * HID * 4, stream);

  k_buildG<<<(CDIM * 196 + 255) / 256, 256, 0, stream>>>(wc, G);
  k_f32_to_bf16<<<4096, 256, 0, stream>>>(x, xb, (size_t)BATCH * CDIM * HW);
  k_transpose_bf16<<<1024, 256, 0, stream>>>(w1, w1t, CDIM, HID);
  k_transpose_bf16<<<1024, 256, 0, stream>>>(w2, w2t, HID, CDIM);
  k_bv<<<CDIM, 256, 0, stream>>>(gb, w2, bv);
  k_build_tt<<<65536, 256, 0, stream>>>(kpe, Tt);
  k_sat<<<BATCH * CDIM, 64, 0, stream>>>(x, SS);

  gemm_bf16<0><<<dim3(32, 24), 256, 0, stream>>>(xb, Tt, y, nullptr,
                                                 3072, 4096, 4096,
                                                 nullptr, nullptr, nullptr);
  k_pc<<<BATCH * CDIM, 256, 0, stream>>>(SS, G, y);
  k_ln<<<1024, 256, 0, stream>>>(y, lnw, lnb, hln);
  gemm_bf16<1><<<dim3(HID / 128, NPIX / 128), 256, 0, stream>>>(
      hln, w1t, nullptr, H1, NPIX, HID, CDIM, b1, nullptr, nullptr);
  k_grn_reduce<<<dim3(HID / 256, BATCH, 16), 256, 0, stream>>>(H1, gx2);
  k_grn_scale<<<BATCH, 256, 0, stream>>>(gx2, gg, sfac);
  k_grn_apply<<<8192, 256, 0, stream>>>(H1, sfac);
  gemm_bf16<2><<<dim3(CDIM / 128, NPIX / 128), 256, 0, stream>>>(
      H1, w2t, out, nullptr, NPIX, CDIM, HID, b2, bv, x);
}
